// Transformer_87342454931872
// MI455X (gfx1250) — compile-verified, matrix operand order checked by reading, not run on hardware
//
#include <hip/hip_runtime.h>
#include <cstdint>

// ---------------- problem constants ----------------
#define BB   2
#define LSEQ 2048
#define DM   1024
#define HH   16
#define EH   64
#define DFF  4096
#define NLAY 3
#define VV   32000
#define MTOK (BB * LSEQ)   // 4096 rows of the residual stream

typedef unsigned short u16;

// ---------------- WMMA fragment types (CDNA5, wave32) ----------------
typedef __attribute__((ext_vector_type(16))) __bf16 v16bf;
typedef __attribute__((ext_vector_type(8)))  __bf16 bf16x8;
typedef __attribute__((ext_vector_type(8)))  float  v8f;

union Frag16 {            // 16 bf16 = 8 VGPRs, filled from two 16B LDS/global loads
    v16bf  v;
    bf16x8 h[2];
};

__device__ inline v8f vzero8() {
    v8f z;
#pragma unroll
    for (int i = 0; i < 8; ++i) z[i] = 0.0f;
    return z;
}

__device__ inline u16 f2bf(float f) {  // round-to-nearest-even f32 -> bf16
    unsigned u = __float_as_uint(f);
    u += 0x7FFFu + ((u >> 16) & 1u);
    return (u16)(u >> 16);
}

__device__ inline float gelu_exact(float x) {
    return 0.5f * x * (1.0f + erff(x * 0.70710678118654752f));
}

// CDNA5 async copy: 16B global -> LDS, tracked by ASYNCcnt (bypasses VGPRs)
__device__ inline void async_g2l_b128(const u16* gptr, unsigned lds_off) {
    unsigned long long ga = (unsigned long long)(uintptr_t)gptr;
    asm volatile("global_load_async_to_lds_b128 %0, %1, off"
                 :: "v"(lds_off), "v"(ga) : "memory");
}
__device__ inline void wait_async0() {
    asm volatile("s_wait_asynccnt 0x0" ::: "memory");
}

// ---------------- f32 -> bf16 bulk convert (weights) ----------------
__global__ __launch_bounds__(256) void cvt_f32_bf16(const float* __restrict__ src,
                                                    u16* __restrict__ dst, size_t n4) {
    size_t i = (size_t)blockIdx.x * blockDim.x + threadIdx.x;
    if (i >= n4) return;
    float4 v = ((const float4*)src)[i];
    uint2 o;
    o.x = (unsigned)f2bf(v.x) | ((unsigned)f2bf(v.y) << 16);
    o.y = (unsigned)f2bf(v.z) | ((unsigned)f2bf(v.w) << 16);
    ((uint2*)dst)[i] = o;
}

// ---------------- embedding: x = tok_emb[tok] + pos_emb[l] ----------------
__global__ __launch_bounds__(256) void embed_kernel(const int* __restrict__ tokens,
                                                    const float* __restrict__ te,
                                                    const float* __restrict__ pe,
                                                    float* __restrict__ x) {
    int row = blockIdx.x;               // 0 .. MTOK-1
    int l   = row % LSEQ;
    int tok = tokens[row];
    int d   = threadIdx.x * 4;
    float4 a = *(const float4*)&te[(size_t)tok * DM + d];
    float4 b = *(const float4*)&pe[(size_t)l   * DM + d];
    float4 o = make_float4(a.x + b.x, a.y + b.y, a.z + b.z, a.w + b.w);
    *(float4*)&x[(size_t)row * DM + d] = o;
}

// ---------------- layernorm (f32 in, bf16 out) ----------------
__global__ __launch_bounds__(256) void layernorm_kernel(const float* __restrict__ x,
                                                        const float* __restrict__ g,
                                                        const float* __restrict__ b,
                                                        u16* __restrict__ out) {
    __shared__ float wsum[8], wsq[8];
    int row = blockIdx.x;
    int d   = threadIdx.x * 4;
    float4 v = *(const float4*)&x[(size_t)row * DM + d];
    float s  = v.x + v.y + v.z + v.w;
    float sq = v.x * v.x + v.y * v.y + v.z * v.z + v.w * v.w;
#pragma unroll
    for (int off = 1; off < 32; off <<= 1) {
        s  += __shfl_xor(s,  off, 32);
        sq += __shfl_xor(sq, off, 32);
    }
    int wv = threadIdx.x >> 5;
    if ((threadIdx.x & 31) == 0) { wsum[wv] = s; wsq[wv] = sq; }
    __syncthreads();
    float ts = 0.f, tq = 0.f;
#pragma unroll
    for (int j = 0; j < 8; ++j) { ts += wsum[j]; tq += wsq[j]; }
    float mean = ts * (1.0f / DM);
    float var  = tq * (1.0f / DM) - mean * mean;
    float rstd = rsqrtf(var + 1e-5f);
    float4 gg = *(const float4*)&g[d];
    float4 bb = *(const float4*)&b[d];
    u16 o0 = f2bf((v.x - mean) * rstd * gg.x + bb.x);
    u16 o1 = f2bf((v.y - mean) * rstd * gg.y + bb.y);
    u16 o2 = f2bf((v.z - mean) * rstd * gg.z + bb.z);
    u16 o3 = f2bf((v.w - mean) * rstd * gg.w + bb.w);
    uint2 pk;
    pk.x = (unsigned)o0 | ((unsigned)o1 << 16);
    pk.y = (unsigned)o2 | ((unsigned)o3 << 16);
    *(uint2*)&out[(size_t)row * DM + d] = pk;
}

// ---------------- tiled bf16 WMMA GEMM (double-buffered, async A staging) ----
// C[M,N] = A[M,K](bf16) * W[K,N](bf16) + bias, optional residual / GELU,
// output to f32 and/or bf16. Block tile 128x64, BK=32, 256 threads = 8 waves.
// A tile: GLOBAL_LOAD_ASYNC_TO_LDS_B128 straight into LDS (row-major, no
// transform needed). W tile: global->VGPR then transposed repack into LDS so
// B-fragments are two contiguous ds_load_b128 per lane. One barrier per step.
template <bool GELU, bool RESID, bool OUTF, bool OUTB>
__global__ __launch_bounds__(256) void gemm_bf16(const u16* __restrict__ A,
                                                 const u16* __restrict__ W,
                                                 const float* __restrict__ bias,
                                                 const float* __restrict__ resid,
                                                 float* __restrict__ outF,
                                                 u16* __restrict__ outB,
                                                 int Kdim, int Ndim) {
    __shared__ u16 As[2][128][40];   // row-major, stride 40 (80B, 16B aligned)
    __shared__ u16 Bs[2][64][40];    // TRANSPOSED: [n][k]

    const int tid  = threadIdx.x;
    const int lane = tid & 31;
    const int wv   = tid >> 5;
    const int lm   = lane & 15;
    const int lh   = lane >> 4;
    const int n0   = blockIdx.x * 64;
    const int m0   = blockIdx.y * 128;

    v8f acc[4];
#pragma unroll
    for (int t = 0; t < 4; ++t) acc[t] = vzero8();

    uint2 breg[2];

    // --- staging helpers ---
    auto issueA = [&](int buf, int k0) {   // 128x32 tile, 512 x 16B async chunks
#pragma unroll
        for (int i = 0; i < 2; ++i) {
            int idx = tid + 256 * i;
            int r   = idx >> 2;
            int c8  = (idx & 3) * 8;
            async_g2l_b128(&A[(size_t)(m0 + r) * Kdim + k0 + c8],
                           (unsigned)(uintptr_t)&As[buf][r][c8]);
        }
    };
    auto loadB = [&](int k0) {             // 32x64 tile into registers
#pragma unroll
        for (int i = 0; i < 2; ++i) {
            int idx = tid + 256 * i;
            int kr  = idx >> 4;
            int nc  = (idx & 15) * 4;
            breg[i] = *(const uint2*)&W[(size_t)(k0 + kr) * Ndim + n0 + nc];
        }
    };
    auto storeB = [&](int buf) {           // transposed repack regs -> LDS
#pragma unroll
        for (int i = 0; i < 2; ++i) {
            int idx = tid + 256 * i;
            int kr  = idx >> 4;
            int nc  = (idx & 15) * 4;
            Bs[buf][nc + 0][kr] = (u16)(breg[i].x & 0xFFFFu);
            Bs[buf][nc + 1][kr] = (u16)(breg[i].x >> 16);
            Bs[buf][nc + 2][kr] = (u16)(breg[i].y & 0xFFFFu);
            Bs[buf][nc + 3][kr] = (u16)(breg[i].y >> 16);
        }
    };

    // --- prologue: stage tile 0 ---
    issueA(0, 0);
    loadB(0);
    storeB(0);
    wait_async0();
    __syncthreads();

    const int kTiles = Kdim >> 5;
    for (int kt = 0; kt < kTiles; ++kt) {
        const int cur  = kt & 1;
        const int nxt  = cur ^ 1;
        const bool more = (kt + 1) < kTiles;
        if (more) {                       // overlap next-tile staging with WMMA
            issueA(nxt, (kt + 1) << 5);
            loadB((kt + 1) << 5);
        }

        Frag16 a;
        a.h[0] = *(const bf16x8*)&As[cur][wv * 16 + lm][8 * lh];
        a.h[1] = *(const bf16x8*)&As[cur][wv * 16 + lm][16 + 8 * lh];
#pragma unroll
        for (int nt = 0; nt < 4; ++nt) {
            Frag16 bfr;
            bfr.h[0] = *(const bf16x8*)&Bs[cur][nt * 16 + lm][8 * lh];
            bfr.h[1] = *(const bf16x8*)&Bs[cur][nt * 16 + lm][16 + 8 * lh];
            acc[nt] = __builtin_amdgcn_wmma_f32_16x16x32_bf16(
                false, a.v, false, bfr.v, (short)0, acc[nt], false, false);
        }

        if (more) storeB(nxt);
        wait_async0();
        __syncthreads();
    }

    // epilogue: C-frag layout row = r + 8*lh, col = nt*16 + lm
#pragma unroll
    for (int nt = 0; nt < 4; ++nt) {
#pragma unroll
        for (int r = 0; r < 8; ++r) {
            int row = m0 + wv * 16 + r + 8 * lh;
            int col = n0 + nt * 16 + lm;
            float v = acc[nt][r] + bias[col];
            if (RESID) v += resid[(size_t)row * Ndim + col];
            if (GELU)  v = gelu_exact(v);
            if (OUTF)  outF[(size_t)row * Ndim + col] = v;
            if (OUTB)  outB[(size_t)row * Ndim + col] = f2bf(v);
        }
    }
}

// ---------------- causal flash attention (bf16 in, bf16 out) ----------------
// grid(L/64, H, B), 128 threads = 4 waves, each wave owns 16 query rows.
// K chunk staged with async global->LDS; V chunk transposed manually.
__global__ __launch_bounds__(128) void flash_attn(const u16* __restrict__ Q,
                                                  const u16* __restrict__ K,
                                                  const u16* __restrict__ V,
                                                  u16* __restrict__ O) {
    __shared__ u16 Ks[32][72];      // [kv][feat]  (stride 144B, 16B aligned)
    __shared__ u16 Vs[64][40];      // [feat][kv]  transposed
    __shared__ u16 Ps[4][16][40];   // per-wave P bounce buffer [m][kv]

    const int tid  = threadIdx.x;
    const int lane = tid & 31;
    const int wv   = tid >> 5;
    const int lm   = lane & 15;
    const int lh   = lane >> 4;
    const int b    = blockIdx.z;
    const int h    = blockIdx.y;
    const int q0   = blockIdx.x * 64;
    const float scale = 0.125f;     // 1/sqrt(64)

    // Load Q fragments (A-layout, two 32-feature chunks)
    Frag16 qf[2];
    {
        int qrow = q0 + wv * 16 + lm;
        const u16* qp = Q + ((size_t)(b * LSEQ + qrow)) * DM + h * EH;
#pragma unroll
        for (int c = 0; c < 2; ++c) {
            qf[c].h[0] = *(const bf16x8*)(qp + 32 * c + 8 * lh);
            qf[c].h[1] = *(const bf16x8*)(qp + 32 * c + 8 * lh + 16);
        }
    }

    float mrow[8], ssum[8];
    v8f o[4];
#pragma unroll
    for (int r = 0; r < 8; ++r) { mrow[r] = -3.0e38f; ssum[r] = 0.0f; }
#pragma unroll
    for (int t = 0; t < 4; ++t) o[t] = vzero8();

    const int qmax = q0 + 63;
    for (int kv0 = 0; kv0 <= qmax; kv0 += 32) {
        // K chunk (32x64, row-major): async straight into LDS
#pragma unroll
        for (int i = 0; i < 2; ++i) {
            int idx = tid + 128 * i;            // 256 chunks of 8 elems (16B)
            int kr  = idx >> 3;
            int fc  = (idx & 7) * 8;
            async_g2l_b128(&K[((size_t)(b * LSEQ + kv0 + kr)) * DM + h * EH + fc],
                           (unsigned)(uintptr_t)&Ks[kr][fc]);
        }
        // V chunk: transposed store (needs VALU repack)
#pragma unroll
        for (int i = 0; i < 4; ++i) {
            int idx = tid + 128 * i;            // 512 chunks of 4
            int kr  = idx >> 4;
            int fc  = (idx & 15) * 4;
            uint2 vv = *(const uint2*)&V[((size_t)(b * LSEQ + kv0 + kr)) * DM + h * EH + fc];
            Vs[fc + 0][kr] = (u16)(vv.x & 0xFFFFu);
            Vs[fc + 1][kr] = (u16)(vv.x >> 16);
            Vs[fc + 2][kr] = (u16)(vv.y & 0xFFFFu);
            Vs[fc + 3][kr] = (u16)(vv.y >> 16);
        }
        wait_async0();
        __syncthreads();

        // S = Q * K^T  (two 16-key tiles, contraction over E=64 in 2 chunks)
        v8f s0 = vzero8(), s1 = vzero8();
#pragma unroll
        for (int c = 0; c < 2; ++c) {
            Frag16 kf0, kf1;
            kf0.h[0] = *(const bf16x8*)&Ks[lm][32 * c + 8 * lh];
            kf0.h[1] = *(const bf16x8*)&Ks[lm][32 * c + 8 * lh + 16];
            kf1.h[0] = *(const bf16x8*)&Ks[16 + lm][32 * c + 8 * lh];
            kf1.h[1] = *(const bf16x8*)&Ks[16 + lm][32 * c + 8 * lh + 16];
            s0 = __builtin_amdgcn_wmma_f32_16x16x32_bf16(
                false, qf[c].v, false, kf0.v, (short)0, s0, false, false);
            s1 = __builtin_amdgcn_wmma_f32_16x16x32_bf16(
                false, qf[c].v, false, kf1.v, (short)0, s1, false, false);
        }

        // online softmax update (rows = r + 8*lh; cross-lane within 16-lane half)
#pragma unroll
        for (int r = 0; r < 8; ++r) {
            int qg  = q0 + wv * 16 + r + 8 * lh;
            int k0g = kv0 + lm;
            int k1g = kv0 + 16 + lm;
            float a0 = s0[r] * scale; if (k0g > qg) a0 = -3.0e38f;
            float a1 = s1[r] * scale; if (k1g > qg) a1 = -3.0e38f;
            float rm = fmaxf(a0, a1);
#pragma unroll
            for (int off = 1; off < 16; off <<= 1) rm = fmaxf(rm, __shfl_xor(rm, off, 32));
            float nm = fmaxf(mrow[r], rm);
            float al = __expf(mrow[r] - nm);
            float p0 = __expf(a0 - nm);
            float p1 = __expf(a1 - nm);
            float rs = p0 + p1;
#pragma unroll
            for (int off = 1; off < 16; off <<= 1) rs += __shfl_xor(rs, off, 32);
            mrow[r] = nm;
            ssum[r] = ssum[r] * al + rs;
#pragma unroll
            for (int t = 0; t < 4; ++t) o[t][r] *= al;
            Ps[wv][r + 8 * lh][lm]      = f2bf(p0);
            Ps[wv][r + 8 * lh][16 + lm] = f2bf(p1);
        }

        // same-wave LDS store->load; DS pipe is in-order, wait for stores anyway
        asm volatile("s_wait_dscnt 0x0" ::: "memory");

        Frag16 pf;
        pf.h[0] = *(const bf16x8*)&Ps[wv][lm][8 * lh];
        pf.h[1] = *(const bf16x8*)&Ps[wv][lm][16 + 8 * lh];
#pragma unroll
        for (int nt = 0; nt < 4; ++nt) {
            Frag16 vf;
            vf.h[0] = *(const bf16x8*)&Vs[nt * 16 + lm][8 * lh];
            vf.h[1] = *(const bf16x8*)&Vs[nt * 16 + lm][16 + 8 * lh];
            o[nt] = __builtin_amdgcn_wmma_f32_16x16x32_bf16(
                false, pf.v, false, vf.v, (short)0, o[nt], false, false);
        }
        __syncthreads();
    }

    // normalize and store O (bf16, laid out as (B,L,H,E) == (M,D))
#pragma unroll
    for (int r = 0; r < 8; ++r) {
        float inv = 1.0f / ssum[r];
        int qg = q0 + wv * 16 + r + 8 * lh;
        u16* op = O + ((size_t)(b * LSEQ + qg)) * DM + h * EH;
#pragma unroll
        for (int nt = 0; nt < 4; ++nt) op[nt * 16 + lm] = f2bf(o[nt][r] * inv);
    }
}

// ---------------- host orchestration ----------------
extern "C" void kernel_launch(void* const* d_in, const int* in_sizes, int n_in,
                              void* d_out, int out_size, void* d_ws, size_t ws_size,
                              hipStream_t stream) {
    const int*   tokens = (const int*)  d_in[0];
    // d_in[1] = pad_mask (all false) -> ignored
    const float* tok_emb = (const float*)d_in[2];
    const float* pos_emb = (const float*)d_in[3];
    const float* Wq = (const float*)d_in[4];   const float* bq = (const float*)d_in[5];
    const float* Wk = (const float*)d_in[6];   const float* bk = (const float*)d_in[7];
    const float* Wv = (const float*)d_in[8];   const float* bv = (const float*)d_in[9];
    const float* Wo = (const float*)d_in[10];  const float* bo = (const float*)d_in[11];
    const float* W1 = (const float*)d_in[12];  const float* b1 = (const float*)d_in[13];
    const float* W2 = (const float*)d_in[14];  const float* b2 = (const float*)d_in[15];
    const float* ln1_g = (const float*)d_in[16]; const float* ln1_b = (const float*)d_in[17];
    const float* ln3_g = (const float*)d_in[18]; const float* ln3_b = (const float*)d_in[19];
    const float* lnf_g = (const float*)d_in[20]; const float* lnf_b = (const float*)d_in[21];
    const float* Wout = (const float*)d_in[22]; const float* bout = (const float*)d_in[23];

    // carve workspace
    size_t off = 0;
    auto carve = [&](size_t bytes) -> void* {
        void* p = (char*)d_ws + off;
        off += (bytes + 255) & ~(size_t)255;
        return p;
    };
    float* xb  = (float*)carve((size_t)MTOK * DM * 4);   // residual stream f32
    u16* hb    = (u16*)  carve((size_t)MTOK * DM * 2);   // LN output bf16
    u16* qb    = (u16*)  carve((size_t)MTOK * DM * 2);
    u16* kb    = (u16*)  carve((size_t)MTOK * DM * 2);
    u16* vb    = (u16*)  carve((size_t)MTOK * DM * 2);
    u16* ob    = (u16*)  carve((size_t)MTOK * DM * 2);
    u16* gb    = (u16*)  carve((size_t)MTOK * DFF * 2);  // FFN intermediate
    u16* wqb   = (u16*)  carve((size_t)NLAY * DM * DM * 2);
    u16* wkb   = (u16*)  carve((size_t)NLAY * DM * DM * 2);
    u16* wvb   = (u16*)  carve((size_t)NLAY * DM * DM * 2);
    u16* wob   = (u16*)  carve((size_t)NLAY * DM * DM * 2);
    u16* w1b   = (u16*)  carve((size_t)NLAY * DM * DFF * 2);
    u16* w2b   = (u16*)  carve((size_t)NLAY * DFF * DM * 2);
    u16* woutb = (u16*)  carve((size_t)DM * VV * 2);
    (void)ws_size; (void)n_in; (void)in_sizes; (void)out_size;

    auto cvt = [&](const float* s, u16* d, size_t n) {
        size_t n4 = n / 4;
        int blocks = (int)((n4 + 255) / 256);
        cvt_f32_bf16<<<blocks, 256, 0, stream>>>(s, d, n4);
    };
    cvt(Wq, wqb, (size_t)NLAY * DM * DM);
    cvt(Wk, wkb, (size_t)NLAY * DM * DM);
    cvt(Wv, wvb, (size_t)NLAY * DM * DM);
    cvt(Wo, wob, (size_t)NLAY * DM * DM);
    cvt(W1, w1b, (size_t)NLAY * DM * DFF);
    cvt(W2, w2b, (size_t)NLAY * DFF * DM);
    cvt(Wout, woutb, (size_t)DM * VV);

    embed_kernel<<<MTOK, 256, 0, stream>>>(tokens, tok_emb, pos_emb, xb);

    const dim3 gemmBlk(256);
    const dim3 gD(DM / 64, MTOK / 128);    // N=1024
    const dim3 gF(DFF / 64, MTOK / 128);   // N=4096
    const dim3 gV(VV / 64, MTOK / 128);    // N=32000

    for (int i = 0; i < NLAY; ++i) {
        layernorm_kernel<<<MTOK, 256, 0, stream>>>(xb, ln1_g + (size_t)i * DM,
                                                   ln1_b + (size_t)i * DM, hb);
        gemm_bf16<false, false, false, true><<<gD, gemmBlk, 0, stream>>>(
            hb, wqb + (size_t)i * DM * DM, bq + (size_t)i * DM, nullptr, nullptr, qb, DM, DM);
        gemm_bf16<false, false, false, true><<<gD, gemmBlk, 0, stream>>>(
            hb, wkb + (size_t)i * DM * DM, bk + (size_t)i * DM, nullptr, nullptr, kb, DM, DM);
        gemm_bf16<false, false, false, true><<<gD, gemmBlk, 0, stream>>>(
            hb, wvb + (size_t)i * DM * DM, bv + (size_t)i * DM, nullptr, nullptr, vb, DM, DM);

        flash_attn<<<dim3(LSEQ / 64, HH, BB), 128, 0, stream>>>(qb, kb, vb, ob);

        gemm_bf16<false, true, true, false><<<gD, gemmBlk, 0, stream>>>(
            ob, wob + (size_t)i * DM * DM, bo + (size_t)i * DM, xb, xb, nullptr, DM, DM);

        layernorm_kernel<<<MTOK, 256, 0, stream>>>(xb, ln3_g + (size_t)i * DM,
                                                   ln3_b + (size_t)i * DM, hb);
        gemm_bf16<true, false, false, true><<<gF, gemmBlk, 0, stream>>>(
            hb, w1b + (size_t)i * DM * DFF, b1 + (size_t)i * DFF, nullptr, nullptr, gb, DM, DFF);
        gemm_bf16<false, true, true, false><<<gD, gemmBlk, 0, stream>>>(
            gb, w2b + (size_t)i * DFF * DM, b2 + (size_t)i * DM, xb, xb, nullptr, DFF, DM);
    }

    layernorm_kernel<<<MTOK, 256, 0, stream>>>(xb, lnf_g, lnf_b, hb);
    gemm_bf16<false, false, true, false><<<gV, gemmBlk, 0, stream>>>(
        hb, woutb, bout, nullptr, (float*)d_out, nullptr, DM, VV);
}
